// StableHierarchicalPooling_89309549953293
// MI455X (gfx1250) — matrix operand order repaired and checked by reading, not compiled, not run on hardware
//
#include <hip/hip_runtime.h>
#include <hip/hip_bf16.h>

typedef __attribute__((ext_vector_type(16))) _Float16 v16h;
typedef __attribute__((ext_vector_type(8)))  _Float16 v8h;
typedef __attribute__((ext_vector_type(8)))  float    v8f;

#define NN 131072
#define CC 128
#define KK 16
#define BBATCH 16
#define TILES (NN/32)          // 4096 tiles of 32 nodes
#define WAVES 4                // waves per block (blockDim = 128)
#define NBLOCKS (TILES/WAVES)  // 1024
#define W1P 136                // padded LDS pitch (halves): +4 banks/row => conflict-free b128
#define HP  136
#define XP  132                // padded LDS pitch (floats) for staged x tile

// d_out layout (concatenated return tuple, all f32):
#define OUT_OFF  0
#define S_OFF    (BBATCH*KK*CC)              // 32768
#define MU_OFF   (S_OFF + NN*KK)             // 2129920
#define LOSS_OFF (MU_OFF + BBATCH*KK*2)      // 2130432

// workspace (floats)
#define WS_AVG  0      // [16]  sum over n of s[n,k]
#define WS_ENT  16     // [1]   sum s*log(s+eps)
#define WS_SUMS 32     // [256] sum_s per (b,k)
#define WS_POS  288    // [512] sum s*pos per (b,k,dim)
#define WS_TOT  800

#define EPSF 1e-9f

__global__ void shp_init(float* __restrict__ out, float* __restrict__ ws) {
    int i = blockIdx.x * blockDim.x + threadIdx.x;
    if (i < BBATCH*KK*CC) out[i] = 0.f;
    if (i < WS_TOT)       ws[i]  = 0.f;
}

__global__ __launch_bounds__(128)
void shp_main(const float* __restrict__ x, const int* __restrict__ batch,
              const float* __restrict__ pos, const float* __restrict__ gum,
              const float* __restrict__ W1, const float* __restrict__ b1,
              const float* __restrict__ W2, const float* __restrict__ b2,
              const float* __restrict__ scaling, const float* __restrict__ amask,
              float* __restrict__ out, float* __restrict__ ws)
{
    __shared__ __attribute__((aligned(16))) _Float16 sW1[CC*W1P];       // 34816 B
    __shared__ __attribute__((aligned(16))) _Float16 sW2[KK*W1P];       //  4352 B
    __shared__ __attribute__((aligned(16))) _Float16 sH[WAVES][16*HP];  // 17408 B
    __shared__ __attribute__((aligned(16))) float    sXf[WAVES][32*XP]; // 67584 B  (124 KB total)

    const int tid  = threadIdx.x;
    const int lane = tid & 31;
    const int wv   = tid >> 5;
    const int hs   = lane >> 4;   // 16-lane half select
    const int lr   = lane & 15;

    const int tile = blockIdx.x * WAVES + wv;   // exactly TILES waves in grid
    const int n0   = tile * 32;

    // ---- async DMA: stage this wave's 32x128 f32 x-tile into LDS (no VGPR round trip).
    // 1024 16B chunks per tile -> 32 async b128 ops per lane, issued before weight staging.
    {
        float* ldsbase = &sXf[wv][0];
        for (int i = 0; i < 32; ++i) {
            const int id    = i * 32 + lane;     // chunk id within tile
            const int row   = id >> 5;           // 32 chunks of 4 floats per row
            const int chunk = id & 31;
            const unsigned int ldsOff =
                (unsigned int)(uintptr_t)(ldsbase + row * XP + chunk * 4); // LDS aperture: addr[31:0] = LDS offset
            const unsigned long long ga =
                (unsigned long long)(uintptr_t)(x + (size_t)(n0 + row) * CC + chunk * 4);
            asm volatile("global_load_async_to_lds_b128 %0, %1, off"
                         :: "v"(ldsOff), "v"(ga) : "memory");
        }
    }

    // ---- stage weights as f16 in LDS (padded pitch) while the x DMA is in flight ----
    for (int i = tid; i < CC*CC; i += 128)
        sW1[(i >> 7) * W1P + (i & 127)] = (_Float16)W1[i];
    for (int i = tid; i < KK*CC; i += 128)
        sW2[(i >> 7) * W1P + (i & 127)] = (_Float16)W2[i];
    __syncthreads();

    // wait for this wave's async tile before touching sXf
    asm volatile("s_wait_asynccnt 0x0" ::: "memory");

    __builtin_prefetch(gum + (size_t)n0 * KK, 0, 1);
    __builtin_prefetch(pos + (size_t)n0 * 2, 0, 1);

    const float sc  = scaling[0];
    const float am  = amask[lr];
    const float b2v = b2[lr];

    float s0[8], s1[8];
    _Float16* hw = &sH[wv][0];
    const float* xw = &sXf[wv][0];

    // ===================== per 16-node subtile: MLP + softmax =====================
    for (int st = 0; st < 2; ++st) {
        const int rbase = n0 + st * 16;

        // A fragments of x from LDS: row = lane&15, two contiguous 8-half K-runs per frag
        v16h ax[4];
        {
            const float* xr = xw + (st*16 + lr) * XP;
            #pragma unroll
            for (int kc = 0; kc < 4; ++kc) {
                const float4 f0 = *(const float4*)(xr + kc*32 + hs*8);
                const float4 f1 = *(const float4*)(xr + kc*32 + hs*8 + 4);
                const float4 f2 = *(const float4*)(xr + kc*32 + 16 + hs*8);
                const float4 f3 = *(const float4*)(xr + kc*32 + 16 + hs*8 + 4);
                v16h a;
                a[0]=(_Float16)f0.x; a[1]=(_Float16)f0.y; a[2]=(_Float16)f0.z; a[3]=(_Float16)f0.w;
                a[4]=(_Float16)f1.x; a[5]=(_Float16)f1.y; a[6]=(_Float16)f1.z; a[7]=(_Float16)f1.w;
                a[8]=(_Float16)f2.x; a[9]=(_Float16)f2.y; a[10]=(_Float16)f2.z; a[11]=(_Float16)f2.w;
                a[12]=(_Float16)f3.x; a[13]=(_Float16)f3.y; a[14]=(_Float16)f3.z; a[15]=(_Float16)f3.w;
                ax[kc] = a;
            }
        }

        // h = relu(x @ W1^T + b1) -> LDS (f16)
        #pragma unroll
        for (int ct = 0; ct < 8; ++ct) {
            const float bias = b1[ct*16 + lr];
            v8f acc;
            #pragma unroll
            for (int r = 0; r < 8; ++r) acc[r] = bias;
            #pragma unroll
            for (int kc = 0; kc < 4; ++kc) {
                const _Float16* wp = &sW1[(ct*16 + lr) * W1P + kc*32 + hs*8];
                v8h blo = *(const v8h*)(wp);
                v8h bhi = *(const v8h*)(wp + 16);
                v16h bf = __builtin_shufflevector(blo, bhi, 0,1,2,3,4,5,6,7,8,9,10,11,12,13,14,15);
                acc = __builtin_amdgcn_wmma_f32_16x16x32_f16(false, ax[kc], false, bf,
                                                             (short)0, acc, false, false);
            }
            #pragma unroll
            for (int r = 0; r < 8; ++r) {
                float v = acc[r] > 0.f ? acc[r] : 0.f;
                hw[(r + hs*8) * HP + ct*16 + lr] = (_Float16)v;
            }
        }

        // logits = h @ W2^T + b2  (single 16-wide output tile)
        v8f lg;
        #pragma unroll
        for (int r = 0; r < 8; ++r) lg[r] = b2v;
        #pragma unroll
        for (int kc = 0; kc < 4; ++kc) {
            const _Float16* hp = hw + lr * HP + kc*32;
            v8h a0 = *(const v8h*)(hp + hs*8);
            v8h a1 = *(const v8h*)(hp + 16 + hs*8);
            v16h af = __builtin_shufflevector(a0, a1, 0,1,2,3,4,5,6,7,8,9,10,11,12,13,14,15);
            const _Float16* wp = &sW2[lr * W1P + kc*32 + hs*8];
            v8h w0 = *(const v8h*)(wp);
            v8h w1 = *(const v8h*)(wp + 16);
            v16h bf = __builtin_shufflevector(w0, w1, 0,1,2,3,4,5,6,7,8,9,10,11,12,13,14,15);
            lg = __builtin_amdgcn_wmma_f32_16x16x32_f16(false, af, false, bf,
                                                        (short)0, lg, false, false);
        }

        // scale, mask, +gumbel, softmax across the 16-lane k dimension
        float* sout = (st == 0) ? s0 : s1;
        #pragma unroll
        for (int r = 0; r < 8; ++r) {
            const int n = rbase + r + hs*8;
            float v = lg[r] * sc;
            v = (am == 0.f) ? -1e9f : v;
            v += gum[(size_t)n * KK + lr];
            float m = v;
            m = fmaxf(m, __shfl_xor(m, 1, 32));
            m = fmaxf(m, __shfl_xor(m, 2, 32));
            m = fmaxf(m, __shfl_xor(m, 4, 32));
            m = fmaxf(m, __shfl_xor(m, 8, 32));
            float e = __expf(v - m);
            float d = e;
            d += __shfl_xor(d, 1, 32);
            d += __shfl_xor(d, 2, 32);
            d += __shfl_xor(d, 4, 32);
            d += __shfl_xor(d, 8, 32);
            float sv = e / d;
            sout[r] = sv;
            out[S_OFF + (size_t)n * KK + lr] = sv;
        }
    }

    // ===================== segmented pooling via WMMA (s^T x) =====================
    int bb0[8], bb1[8];
    #pragma unroll
    for (int j = 0; j < 8; ++j) {
        bb0[j] = batch[n0 + hs*8 + j];
        bb1[j] = batch[n0 + 16 + hs*8 + j];
    }
    const int bfirst = batch[n0];
    const int blast  = batch[n0 + 31];

    _Float16 s0h[8], s1h[8];
    #pragma unroll
    for (int j = 0; j < 8; ++j) { s0h[j] = (_Float16)s0[j]; s1h[j] = (_Float16)s1[j]; }

    for (int bc = bfirst; bc <= blast; ++bc) {   // uniform per wave (batch sorted)
        v16h af;   // A = s^T: row=k=lane&15, K-dim=node (matches C/D register layout!)
        #pragma unroll
        for (int j = 0; j < 8; ++j) {
            af[j]     = (bb0[j] == bc) ? s0h[j] : (_Float16)0.f;
            af[8 + j] = (bb1[j] == bc) ? s1h[j] : (_Float16)0.f;
        }
        #pragma unroll
        for (int ct = 0; ct < 8; ++ct) {
            const int c = ct*16 + lr;
            v16h bf;   // B = x (32 nodes x 16 cols), transposed gather from LDS-staged tile
            #pragma unroll
            for (int j = 0; j < 16; ++j)
                bf[j] = (_Float16)xw[((j & 8)*2 + hs*8 + (j & 7)) * XP + c];
            v8f acc;
            #pragma unroll
            for (int r = 0; r < 8; ++r) acc[r] = 0.f;
            acc = __builtin_amdgcn_wmma_f32_16x16x32_f16(false, af, false, bf,
                                                         (short)0, acc, false, false);
            #pragma unroll
            for (int r = 0; r < 8; ++r)
                atomicAdd(&out[((size_t)bc*KK + r + hs*8) * CC + c], acc[r]);
        }
    }

    // ===================== scalar reductions (avg_s, entropy, mu sums) ============
    float ksum = 0.f, ent = 0.f;
    float ax_ = 0.f, ay_ = 0.f, as_ = 0.f;
    int curb = bb0[0];
    #pragma unroll
    for (int j = 0; j < 16; ++j) {
        const int nloc = (j < 8) ? (hs*8 + j) : (16 + hs*8 + (j - 8));
        const int bn   = (j < 8) ? bb0[j] : bb1[j - 8];
        const float sv = (j < 8) ? s0[j] : s1[j - 8];
        const int n = n0 + nloc;
        ksum += sv;
        ent  += sv * __logf(sv + EPSF);
        if (bn != curb) {
            atomicAdd(&ws[WS_POS + (curb*KK + lr)*2 + 0], ax_);
            atomicAdd(&ws[WS_POS + (curb*KK + lr)*2 + 1], ay_);
            atomicAdd(&ws[WS_SUMS + curb*KK + lr], as_);
            curb = bn; ax_ = ay_ = as_ = 0.f;
        }
        ax_ += sv * pos[2*(size_t)n + 0];
        ay_ += sv * pos[2*(size_t)n + 1];
        as_ += sv;
    }
    atomicAdd(&ws[WS_POS + (curb*KK + lr)*2 + 0], ax_);
    atomicAdd(&ws[WS_POS + (curb*KK + lr)*2 + 1], ay_);
    atomicAdd(&ws[WS_SUMS + curb*KK + lr], as_);

    atomicAdd(&ws[WS_AVG + lr], ksum);
    ent += __shfl_xor(ent, 1, 32);
    ent += __shfl_xor(ent, 2, 32);
    ent += __shfl_xor(ent, 4, 32);
    ent += __shfl_xor(ent, 8, 32);
    ent += __shfl_xor(ent, 16, 32);
    if (lane == 0) atomicAdd(&ws[WS_ENT], ent);
}

__global__ __launch_bounds__(256)
void shp_final(const float* __restrict__ amask, float* __restrict__ out,
               const float* __restrict__ ws)
{
    __shared__ float savg[KK];
    __shared__ float smu[BBATCH*KK*2];
    __shared__ float ssep[256];
    const int tid = threadIdx.x;
    if (tid < KK) savg[tid] = ws[WS_AVG + tid] * (1.f / (float)NN);
    for (int i = tid; i < BBATCH*KK*2; i += 256) {
        float m = ws[WS_POS + i] / (ws[WS_SUMS + (i >> 1)] + EPSF);
        smu[i] = m;
        out[MU_OFF + i] = m;
    }
    __syncthreads();
    {   // separation: pairwise repulsion between super-node centers
        const int b = tid >> 4, k1 = tid & 15;
        const float mx = smu[(b*KK + k1)*2], my = smu[(b*KK + k1)*2 + 1];
        float acc = 0.f;
        for (int k2 = 0; k2 < KK; ++k2) {
            if (k2 == k1) continue;
            float dx = mx - smu[(b*KK + k2)*2];
            float dy = my - smu[(b*KK + k2)*2 + 1];
            acc += 1.f / (dx*dx + dy*dy + 1.f);
        }
        ssep[tid] = acc;
    }
    __syncthreads();
    if (tid == 0) {
        float sep = 0.f;
        for (int i = 0; i < 256; ++i) sep += ssep[i];
        sep *= 1.f / (float)(KK * (KK - 1));
        const float p = 1.f / (float)KK;
        float entropy = -ws[WS_ENT] * (1.f / (float)NN);
        float diversity = 0.f, prun = 0.f, asum = 0.f, coll = 0.f, amean = 0.f;
        for (int k = 0; k < KK; ++k) {
            float a = savg[k];
            diversity += p * __logf(p / (a + EPSF));
            prun  += fabsf(a * (1.f - amask[k]));
            asum  += amask[k];
            coll  += (a - p) * (a - p);
            amean += a;
        }
        prun *= p; amean *= p;
        float var = 0.f;
        for (int k = 0; k < KK; ++k) { float d = savg[k] - amean; var += d * d; }
        var *= p;
        out[LOSS_OFF + 0] = entropy;
        out[LOSS_OFF + 1] = diversity;
        out[LOSS_OFF + 2] = 0.f;                 // spatial
        out[LOSS_OFF + 3] = prun;
        out[LOSS_OFF + 4] = asum * p * 0.01f;    // sparsity
        out[LOSS_OFF + 5] = 0.f;                 // temporal * W
        out[LOSS_OFF + 6] = coll * 2.f;          // collapse
        out[LOSS_OFF + 7] = sqrtf(var);          // balance (population std)
        out[LOSS_OFF + 8] = sep;
    }
}

extern "C" void kernel_launch(void* const* d_in, const int* in_sizes, int n_in,
                              void* d_out, int out_size, void* d_ws, size_t ws_size,
                              hipStream_t stream) {
    const float* x       = (const float*)d_in[0];
    const int*   batch   = (const int*)d_in[1];   // int32 (jax x64 disabled)
    const float* pos     = (const float*)d_in[2];
    const float* gum     = (const float*)d_in[3];
    const float* W1      = (const float*)d_in[4];
    const float* b1      = (const float*)d_in[5];
    const float* W2      = (const float*)d_in[6];
    const float* b2      = (const float*)d_in[7];
    const float* scaling = (const float*)d_in[8];
    const float* amask   = (const float*)d_in[9];
    float* out = (float*)d_out;
    float* ws  = (float*)d_ws;

    shp_init<<<(BBATCH*KK*CC + 255) / 256, 256, 0, stream>>>(out, ws);
    shp_main<<<NBLOCKS, 128, 0, stream>>>(x, batch, pos, gum, W1, b1, W2, b2,
                                          scaling, amask, out, ws);
    shp_final<<<1, 256, 0, stream>>>(amask, out, ws);
}
